// SelfAttention_78391743087291
// MI455X (gfx1250) — matrix operand order repaired and checked by reading, not compiled, not run on hardware
//
#include <hip/hip_runtime.h>

typedef __attribute__((ext_vector_type(16))) _Float16 v16h;
typedef __attribute__((ext_vector_type(8)))  float    v8f;
typedef __attribute__((ext_vector_type(4)))  float    v4f;

union F16x16 { v16h h; v4f q[2]; };

#define SWZ_XOR_F(v, m) __int_as_float(__builtin_amdgcn_ds_swizzle(__float_as_int(v), (((m) << 10) | 0x1f)))

// Problem constants
constexpr int  Bb   = 2;
constexpr int  Tt   = 2048;
constexpr int  Cc   = 2048;
constexpr int  NGrp = 4;
constexpr int  NHd  = 16;
constexpr int  HSz  = 128;
constexpr int  Mrows = Bb * Tt;                 // 4096
constexpr int  Nqkv  = (NHd + 2 * NGrp) * HSz;  // 3072
// softmax scale folded into q: (1/sqrt(128)) * log2(e)  -> exp2-domain softmax
constexpr float QSCALE = 0.088388347648318447f * 1.4426950408889634f;

// Workspace layout (bytes)
constexpr size_t OFF_XH  = 0;
constexpr size_t SZ_XH   = (size_t)Mrows * Cc * 2;
constexpr size_t OFF_WA  = OFF_XH + SZ_XH;
constexpr size_t SZ_WA   = (size_t)Cc * Nqkv * 2;
constexpr size_t OFF_WP  = OFF_WA + SZ_WA;
constexpr size_t SZ_WP   = (size_t)Cc * Cc * 2;
constexpr size_t OFF_QKV = OFF_WP + SZ_WP;
constexpr size_t SZ_QKV  = (size_t)Mrows * Nqkv * 2;
constexpr size_t OFF_KT  = OFF_QKV + SZ_QKV;
constexpr size_t SZ_KT   = (size_t)Bb * NGrp * HSz * Tt * 2;
constexpr size_t OFF_YH  = OFF_KT + SZ_KT;

// ---------------------------------------------------------------- f32 -> f16
__global__ void cvt_f32_to_f16(const float* __restrict__ src,
                               _Float16* __restrict__ dst, int n) {
  for (int i = blockIdx.x * blockDim.x + threadIdx.x; i < n;
       i += gridDim.x * blockDim.x)
    dst[i] = (_Float16)src[i];
}

// ---------------------------------------------------------------- async tile fill
// Per thread: 2 x b128 for the A tile (32B), 4 x b128 for the B tile (64B).
// INST_OFFSET is added to BOTH the LDS and global addresses (ISA 8.x async pseudocode),
// so one (lds,vaddr) pair + immediate offsets covers contiguous chunks.
__device__ __forceinline__ void async_fill(unsigned ldsA, unsigned offA,
                                           const _Float16* A,
                                           unsigned ldsB, unsigned offB,
                                           const _Float16* Bm) {
  asm volatile(
      "global_load_async_to_lds_b128 %0, %1, %4\n\t"
      "global_load_async_to_lds_b128 %0, %1, %4 offset:16\n\t"
      "global_load_async_to_lds_b128 %2, %3, %5\n\t"
      "global_load_async_to_lds_b128 %2, %3, %5 offset:16\n\t"
      "global_load_async_to_lds_b128 %2, %3, %5 offset:32\n\t"
      "global_load_async_to_lds_b128 %2, %3, %5 offset:48\n\t"
      :
      : "v"(ldsA), "v"(offA), "v"(ldsB), "v"(offB), "s"(A), "s"(Bm)
      : "memory");
}

// ---------------------------------------------------------------- GEMM (f16 in, WMMA, f16/f32 out)
// C[M,N] = A[M,K] * B[K,N], A stride K, B stride N.
// Block: 128 threads = 4 waves; tile 64(M) x 128(N), BK = 32, double-buffered LDS
// filled with CDNA5 async global->LDS copies (ASYNCcnt pipelined).
template <bool OUT_F32>
__global__ __launch_bounds__(128, 1) void gemm_f16_wmma(
    const _Float16* __restrict__ A, const _Float16* __restrict__ Bm,
    void* __restrict__ Cout, int M, int N, int K) {
  constexpr int BM = 64, BN = 128, BK = 32;
  __shared__ __align__(16) _Float16 Ah[2][BM * BK];  // row-major, stride BK
  __shared__ __align__(16) _Float16 Bh[2][BK * BN];  // row-major, stride BN

  const int tid  = threadIdx.x;
  const int wave = tid >> 5;
  const int lane = tid & 31;
  const int m0   = blockIdx.y * BM;
  const int n0   = blockIdx.x * BN;

  // fill assignments: A = 16 halfs/thread, B = 32 halfs/thread
  const int rowA = tid >> 1, colA = (tid & 1) * 16;
  const int rowB = tid >> 2, colB = (tid & 3) * 32;
  const unsigned ldsA0 = (unsigned)(size_t)(&Ah[0][rowA * BK + colA]);
  const unsigned ldsA1 = (unsigned)(size_t)(&Ah[1][rowA * BK + colA]);
  const unsigned ldsB0 = (unsigned)(size_t)(&Bh[0][rowB * BN + colB]);
  const unsigned ldsB1 = (unsigned)(size_t)(&Bh[1][rowB * BN + colB]);
  const unsigned baseA = (unsigned)(((size_t)(m0 + rowA) * K + colA) * 2);
  const unsigned baseB = (unsigned)(((size_t)rowB * N + n0 + colB) * 2);

  v8f acc[8] = {};

  const int mrow = wave * 16 + (lane & 15);
  const int klo  = (lane < 16) ? 0 : 8;
  const int nIter = K / BK;

  // prologue: prefetch tile 0 into buffer 0
  async_fill(ldsA0, baseA, A, ldsB0, baseB, Bm);

  for (int i = 0; i < nIter; ++i) {
    const int cur = i & 1;
    if (i + 1 < nIter) {
      // prefetch next tile into the other buffer, then wait for current tile
      async_fill(cur ? ldsA0 : ldsA1,
                 baseA + (unsigned)((i + 1) * BK * 2),
                 A,
                 cur ? ldsB0 : ldsB1,
                 baseB + (unsigned)((size_t)(i + 1) * BK * N * 2),
                 Bm);
      asm volatile("s_wait_asynccnt 0x6" ::: "memory");
    } else {
      asm volatile("s_wait_asynccnt 0x0" ::: "memory");
    }
    __syncthreads();

    // A fragment (16x32): lane<16 holds K {0..7,16..23}, lane>=16 {8..15,24..31}
    F16x16 fa;
    fa.q[0] = *(const v4f*)(&Ah[cur][mrow * BK + klo]);
    fa.q[1] = *(const v4f*)(&Ah[cur][mrow * BK + klo + 16]);

#pragma unroll
    for (int j = 0; j < 8; ++j) {
      // B fragment (32x16): lane%16 = N column, elements = K in same order as A
      F16x16 fb;
#pragma unroll
      for (int s = 0; s < 16; ++s)
        fb.h[s] = Bh[cur][(klo + (s < 8 ? s : s + 8)) * BN + j * 16 + (lane & 15)];
      acc[j] = __builtin_amdgcn_wmma_f32_16x16x32_f16(
          false, fa.h, false, fb.h, (short)0, acc[j], false, false);
    }
    __syncthreads();
  }

  // C layout: vgpr r, lane -> (m = r + (lane>>4)*8, n = lane&15)
  const int mo = m0 + wave * 16 + (lane >> 4) * 8;
  const int no = n0 + (lane & 15);
#pragma unroll
  for (int j = 0; j < 8; ++j)
#pragma unroll
    for (int r = 0; r < 8; ++r) {
      size_t idx = (size_t)(mo + r) * N + (no + j * 16);
      if constexpr (OUT_F32)
        ((float*)Cout)[idx] = acc[j][r];
      else
        ((_Float16*)Cout)[idx] = (_Float16)acc[j][r];
    }
}

// ---------------------------------------------------------------- RoPE (+ scale fold, + k transpose)
// qkvh layout: [b][t][g][j][d], j: 0..3 = q heads, 4 = k, 5 = v. In-place for q,
// k written rotated to kT[b][g][d][t].
__global__ void rope_kernel(_Float16* __restrict__ qkvh,
                            _Float16* __restrict__ kT,
                            const float* __restrict__ cosT,
                            const float* __restrict__ sinT) {
  size_t id = (size_t)blockIdx.x * blockDim.x + threadIdx.x;
  const int d    = (int)(id & 63);
  const int slot = (int)((id >> 6) % 5);
  size_t rem = id / (64 * 5);
  const int g = (int)(rem % NGrp);
  size_t bt = rem / NGrp;            // b*T + t
  const int t = (int)(bt % Tt);

  size_t base = bt * Nqkv + (size_t)g * (6 * HSz) + (size_t)slot * HSz;
  float x1 = (float)qkvh[base + d];
  float x2 = (float)qkvh[base + d + 64];
  float c = cosT[t * 64 + d];
  float s = sinT[t * 64 + d];
  float o1 = x1 * c - x2 * s;
  float o2 = x1 * s + x2 * c;

  if (slot < 4) {                    // q: scaled, in place
    qkvh[base + d]      = (_Float16)(o1 * QSCALE);
    qkvh[base + d + 64] = (_Float16)(o2 * QSCALE);
  } else {                           // k: transposed out
    const int b = (int)(bt / Tt);
    size_t kb = ((size_t)(b * NGrp + g) * HSz) * Tt + t;
    kT[kb + (size_t)d * Tt]        = (_Float16)o1;
    kT[kb + (size_t)(d + 64) * Tt] = (_Float16)o2;
  }
}

// ---------------------------------------------------------------- Flash attention
// Grid: (T/16, NGrp, Bb); 128 threads = 4 waves, wave w = q-head p within group.
__global__ __launch_bounds__(128, 1) void attn_kernel(
    const _Float16* __restrict__ qkvh, const _Float16* __restrict__ kT,
    _Float16* __restrict__ yh) {
  const int qt = blockIdx.x, g = blockIdx.y, b = blockIdx.z;
  const int wave = threadIdx.x >> 5;   // = p (q head within group)
  const int lane = threadIdx.x & 31;
  const int m  = lane & 15;
  const int hi = lane >> 4;

  __shared__ __align__(16) _Float16 Pl[4][16 * 32];  // per-wave P tile, row-major stride 32

  // q fragments (16 rows x 128, scale pre-folded), kept in registers
  F16x16 qa[4];
  const size_t qrow =
      ((size_t)b * Tt + (size_t)qt * 16 + m) * Nqkv + (size_t)g * (6 * HSz) + (size_t)wave * HSz;
#pragma unroll
  for (int c = 0; c < 4; ++c) {
    const _Float16* p = qkvh + qrow + c * 32 + hi * 8;
    qa[c].q[0] = *(const v4f*)(p);
    qa[c].q[1] = *(const v4f*)(p + 16);
  }

  v8f acc[8] = {};
  float rowM[8], rowL[8];
#pragma unroll
  for (int r = 0; r < 8; ++r) { rowM[r] = -3.0e38f; rowL[r] = 0.0f; }

  const size_t kTbase = ((size_t)(b * NGrp + g) * HSz) * Tt;
  const size_t vbase  = ((size_t)b * Tt) * Nqkv + (size_t)g * (6 * HSz) + 5 * HSz;

  for (int kt = 0; kt < Tt; kt += 32) {
    // ---- scores: two 16x16 tiles over 32 keys, accumulate over d=128
    v8f s0 = {}, s1 = {};
#pragma unroll
    for (int c = 0; c < 4; ++c) {
      const _Float16* pk = kT + kTbase + (size_t)(c * 32) * Tt + kt + m;
      F16x16 b0, b1;
#pragma unroll
      for (int s = 0; s < 16; ++s) {
        const int kd = hi * 8 + (s < 8 ? s : s + 8);
        b0.h[s] = pk[(size_t)kd * Tt];
        b1.h[s] = pk[(size_t)kd * Tt + 16];
      }
      s0 = __builtin_amdgcn_wmma_f32_16x16x32_f16(false, qa[c].h, false, b0.h,
                                                  (short)0, s0, false, false);
      s1 = __builtin_amdgcn_wmma_f32_16x16x32_f16(false, qa[c].h, false, b1.h,
                                                  (short)0, s1, false, false);
    }

    // ---- online softmax (exp2 domain; scale already folded into q)
#pragma unroll
    for (int r = 0; r < 8; ++r) {
      float mx = fmaxf(s0[r], s1[r]);
      mx = fmaxf(mx, SWZ_XOR_F(mx, 1));
      mx = fmaxf(mx, SWZ_XOR_F(mx, 2));
      mx = fmaxf(mx, SWZ_XOR_F(mx, 4));
      mx = fmaxf(mx, SWZ_XOR_F(mx, 8));
      float newM = fmaxf(rowM[r], mx);
      float corr = __builtin_amdgcn_exp2f(rowM[r] - newM);
      rowM[r] = newM;
      float e0 = __builtin_amdgcn_exp2f(s0[r] - newM);
      float e1 = __builtin_amdgcn_exp2f(s1[r] - newM);
      float sum = e0 + e1;
      sum += SWZ_XOR_F(sum, 1);
      sum += SWZ_XOR_F(sum, 2);
      sum += SWZ_XOR_F(sum, 4);
      sum += SWZ_XOR_F(sum, 8);
      rowL[r] = rowL[r] * corr + sum;
#pragma unroll
      for (int j = 0; j < 8; ++j) acc[j][r] *= corr;
      // stage P tile (f16) to LDS: row = m-index, col = key offset within 32
      Pl[wave][(r + hi * 8) * 32 + (lane & 15)]      = (_Float16)e0;
      Pl[wave][(r + hi * 8) * 32 + (lane & 15) + 16] = (_Float16)e1;
    }
    asm volatile("s_wait_dscnt 0" ::: "memory");

    // reload P as an A-fragment (16x32)
    F16x16 pf;
    const _Float16* pp = &Pl[wave][m * 32 + hi * 8];
    pf.q[0] = *(const v4f*)(pp);
    pf.q[1] = *(const v4f*)(pp + 16);

    // ---- y += P @ V  (8 d-tiles of 16)
#pragma unroll
    for (int j = 0; j < 8; ++j) {
      const _Float16* pv = qkvh + vbase + (size_t)kt * Nqkv + j * 16 + m;
      F16x16 vb;
#pragma unroll
      for (int s = 0; s < 16; ++s)
        vb.h[s] = pv[(size_t)(hi * 8 + (s < 8 ? s : s + 8)) * Nqkv];
      acc[j] = __builtin_amdgcn_wmma_f32_16x16x32_f16(false, pf.h, false, vb.h,
                                                      (short)0, acc[j], false, false);
    }
  }

  // ---- epilogue: normalize, write y as f16 [b][t][h][d], h = g*4 + wave
#pragma unroll
  for (int j = 0; j < 8; ++j)
#pragma unroll
    for (int r = 0; r < 8; ++r) {
      float y = acc[j][r] / rowL[r];
      size_t row = (size_t)b * Tt + (size_t)qt * 16 + r + hi * 8;
      size_t col = (size_t)(g * 4 + wave) * HSz + j * 16 + (lane & 15);
      yh[row * Cc + col] = (_Float16)y;
    }
}

// ---------------------------------------------------------------- launch
extern "C" void kernel_launch(void* const* d_in, const int* in_sizes, int n_in,
                              void* d_out, int out_size, void* d_ws, size_t ws_size,
                              hipStream_t stream) {
  const float* x      = (const float*)d_in[0];
  const float* cosT   = (const float*)d_in[1];
  const float* sinT   = (const float*)d_in[2];
  const float* w_attn = (const float*)d_in[3];
  const float* w_proj = (const float*)d_in[4];

  char* ws = (char*)d_ws;
  _Float16* xh   = (_Float16*)(ws + OFF_XH);
  _Float16* wAh  = (_Float16*)(ws + OFF_WA);
  _Float16* wPh  = (_Float16*)(ws + OFF_WP);
  _Float16* qkvh = (_Float16*)(ws + OFF_QKV);
  _Float16* kTh  = (_Float16*)(ws + OFF_KT);
  _Float16* yh   = (_Float16*)(ws + OFF_YH);

  cvt_f32_to_f16<<<4096, 256, 0, stream>>>(x, xh, Mrows * Cc);
  cvt_f32_to_f16<<<4096, 256, 0, stream>>>(w_attn, wAh, Cc * Nqkv);
  cvt_f32_to_f16<<<4096, 256, 0, stream>>>(w_proj, wPh, Cc * Cc);

  gemm_f16_wmma<false><<<dim3(Nqkv / 128, Mrows / 64), 128, 0, stream>>>(
      xh, wAh, (void*)qkvh, Mrows, Nqkv, Cc);

  rope_kernel<<<(Bb * Tt * NGrp * 5 * 64) / 256, 256, 0, stream>>>(
      qkvh, kTh, cosT, sinT);

  attn_kernel<<<dim3(Tt / 16, NGrp, Bb), 128, 0, stream>>>(qkvh, kTh, yh);

  gemm_f16_wmma<true><<<dim3(Cc / 128, Mrows / 64), 128, 0, stream>>>(
      yh, wPh, d_out, Mrows, Cc, Cc);
}
